// EnhancedGATModel_45947560133209
// MI455X (gfx1250) — matrix operand order repaired and checked by reading, not compiled, not run on hardware
//
#include <hip/hip_runtime.h>
#include <cstdint>
#include <cstddef>

// ---------------- problem constants ----------------
constexpr int kN  = 102400;
constexpr int kE  = 307200;
constexpr int kB  = 1024;
constexpr int kIN = 78;
constexpr int kED = 9;
constexpr int kH  = 256;
constexpr int kNH = 4;
constexpr int kDH = 64;
constexpr int kL  = 3;
constexpr float kEPS = 1e-5f;

typedef __attribute__((ext_vector_type(16))) _Float16 v16h;
typedef __attribute__((ext_vector_type(4)))  _Float16 v4h;
typedef __attribute__((ext_vector_type(8)))  float    v8f;

#define DEVFN __device__ __forceinline__

// order-preserving float<->uint map (for atomicMax on floats of any sign)
DEVFN unsigned ford(float f) {
    unsigned u = __float_as_uint(f);
    return (u & 0x80000000u) ? ~u : (u | 0x80000000u);
}
DEVFN float funord(unsigned u) {
    return (u & 0x80000000u) ? __uint_as_float(u ^ 0x80000000u) : __uint_as_float(~u);
}

DEVFN float apply_act(int A, float v) {
    if (A == 1) return v > 0.f ? v : expm1f(v);     // ELU
    if (A == 2) return fmaxf(v, 0.f);               // ReLU
    return v;
}

// A-tile k -> (lane-half, fragment index) permutation per CDNA5 16-bit A layout:
// k in [0,8): half 0, i=k ; [8,16): half 1, i=k-8 ; [16,24): half 0, i=k-8 ; [24,32): half 1, i=k-16
DEVFN int a_half(int k)  { return (k >> 3) & 1; }
DEVFN int a_idx(int k)   { return (k & 7) + ((k >> 4) << 3); }

// ---------------- WMMA GEMM: Y[M,Nout] = act(X[M,K] @ W[Nout,K]^T + b) ----------------
// REQUIRES (guaranteed by all call sites): M % 64 == 0, Nout % 128 == 0.
// block = 256 threads (8 waves); block tile = 64 rows x 128 cols; K staged by 32.
// Each wave owns one 16-col tile and all 4 row tiles -> 4 WMMAs per chunk, B fragment reused.
// LDS tiles stored in WMMA fragment order: each lane reads its fragment as 32B contiguous.
template<int ACT>
__global__ void __launch_bounds__(256)
gemm_bias_act(const float* __restrict__ X, int ldx,
              const float* __restrict__ W,      // [Nout, K] row-major
              const float* __restrict__ bias,   // [Nout]
              float* __restrict__ Y, int ldy,
              int M, int K, int Nout)
{
    __shared__ __align__(64) _Float16 lA[4][32][16];  // [row-tile][lane][i] fragment layout
    __shared__ __align__(64) _Float16 lB[128][32];    // [n][16*a_half(k)+a_idx(k)] (== B fragment layout)

    const int m0   = blockIdx.x * 64;
    const int n0   = blockIdx.y * 128;
    const int t    = threadIdx.x;
    const int wave = t >> 5;
    const int lane = t & 31;
    const int half = lane >> 4;
    const int mr   = lane & 15;

    const bool vecOK = ((ldx & 3) == 0) && ((K & 3) == 0);
    const int  kfull = vecOK ? (K & ~31) : 0;

    v8f acc0 = {}, acc1 = {}, acc2 = {}, acc3 = {};

    auto compute = [&]() {
        __syncthreads();
        v16h b  = *(const v16h*)&lB[wave * 16 + mr][half * 16];   // 2 x ds_load_b128
        v16h a0 = *(const v16h*)&lA[0][lane][0];
        v16h a1 = *(const v16h*)&lA[1][lane][0];
        v16h a2 = *(const v16h*)&lA[2][lane][0];
        v16h a3 = *(const v16h*)&lA[3][lane][0];
        acc0 = __builtin_amdgcn_wmma_f32_16x16x32_f16(false, a0, false, b, (short)0, acc0, false, false);
        acc1 = __builtin_amdgcn_wmma_f32_16x16x32_f16(false, a1, false, b, (short)0, acc1, false, false);
        acc2 = __builtin_amdgcn_wmma_f32_16x16x32_f16(false, a2, false, b, (short)0, acc2, false, false);
        acc3 = __builtin_amdgcn_wmma_f32_16x16x32_f16(false, a3, false, b, (short)0, acc3, false, false);
        __syncthreads();
    };

    int k0 = 0;
    // ---- main loop: float4 global loads, packed half stores, no guards ----
    for (; k0 < kfull; k0 += 32) {
        #pragma unroll
        for (int j = 0; j < 2; ++j) {        // A tile: 64 rows x 32 k = 512 float4
            int idx4 = t + 256 * j;
            int r = idx4 >> 3, k = (idx4 & 7) * 4;
            const float* px = X + (size_t)(m0 + r) * ldx + k0 + k;
            float4 f = *(const float4*)px;
            if (k0 + 64 <= K) __builtin_prefetch(px + 32, 0, 3);
            v4h h;
            h[0] = (_Float16)f.x; h[1] = (_Float16)f.y;
            h[2] = (_Float16)f.z; h[3] = (_Float16)f.w;
            *(v4h*)&lA[r >> 4][(r & 15) + 16 * a_half(k)][a_idx(k)] = h;
        }
        #pragma unroll
        for (int j = 0; j < 4; ++j) {        // B tile: 128 n x 32 k = 1024 float4
            int idx4 = t + 256 * j;
            int n = idx4 >> 3, k = (idx4 & 7) * 4;
            const float* pw = W + (size_t)(n0 + n) * K + k0 + k;
            float4 f = *(const float4*)pw;
            if (k0 + 64 <= K) __builtin_prefetch(pw + 32, 0, 3);
            v4h h;
            h[0] = (_Float16)f.x; h[1] = (_Float16)f.y;
            h[2] = (_Float16)f.z; h[3] = (_Float16)f.w;
            *(v4h*)&lB[n][16 * a_half(k) + a_idx(k)] = h;
        }
        compute();
    }
    // ---- tail: clamped index + select, branch-free scalar loads ----
    for (; k0 < K; k0 += 32) {
        for (int idx = t; idx < 2048; idx += 256) {
            int r = idx >> 5, k = idx & 31;
            int kg = k0 + k;
            int kc = kg < K ? kg : K - 1;
            float v = X[(size_t)(m0 + r) * ldx + kc];
            v = (kg < K) ? v : 0.f;
            lA[r >> 4][(r & 15) + 16 * a_half(k)][a_idx(k)] = (_Float16)v;
        }
        for (int idx = t; idx < 4096; idx += 256) {
            int n = idx >> 5, k = idx & 31;
            int kg = k0 + k;
            int kc = kg < K ? kg : K - 1;
            float v = W[(size_t)(n0 + n) * K + kc];
            v = (kg < K) ? v : 0.f;
            lB[n][16 * a_half(k) + a_idx(k)] = (_Float16)v;
        }
        compute();
    }

    // C/D layout: lane -> N col (lane%16), vgpr r -> M row = r + 8*(lane/16)
    const int ncol = n0 + wave * 16 + mr;
    const float bv = bias[ncol];
    #pragma unroll
    for (int r = 0; r < 8; ++r) {
        int mrow = m0 + r + 8 * half;
        Y[(size_t)(mrow +  0) * ldy + ncol] = apply_act(ACT, acc0[r] + bv);
        Y[(size_t)(mrow + 16) * ldy + ncol] = apply_act(ACT, acc1[r] + bv);
        Y[(size_t)(mrow + 32) * ldy + ncol] = apply_act(ACT, acc2[r] + bv);
        Y[(size_t)(mrow + 48) * ldy + ncol] = apply_act(ACT, acc3[r] + bv);
    }
}

static void gemm(int act, const float* X, int ldx, const float* W, const float* bias,
                 float* Y, int ldy, int M, int K, int Nout, hipStream_t s)
{
    dim3 g((M + 63) / 64, (Nout + 127) / 128), b(256);
    if (act == 1)      gemm_bias_act<1><<<g, b, 0, s>>>(X, ldx, W, bias, Y, ldy, M, K, Nout);
    else if (act == 2) gemm_bias_act<2><<<g, b, 0, s>>>(X, ldx, W, bias, Y, ldy, M, K, Nout);
    else               gemm_bias_act<0><<<g, b, 0, s>>>(X, ldx, W, bias, Y, ldy, M, K, Nout);
}

// ---------------- small helper kernels ----------------
__global__ void k_count_deg(const int* __restrict__ dst, float* deg, int E) {
    int i = blockIdx.x * blockDim.x + threadIdx.x;
    if (i < E) atomicAdd(&deg[dst[i]], 1.f);
}
__global__ void k_invdeg(const float* deg, float* inv, int n) {
    int i = blockIdx.x * blockDim.x + threadIdx.x;
    if (i < n) inv[i] = 1.f / fmaxf(deg[i], 1.f);
}
__global__ void k_count_int(const int* __restrict__ n2g, int* cnt, int n) {
    int i = blockIdx.x * blockDim.x + threadIdx.x;
    if (i < n) atomicAdd(&cnt[n2g[i]], 1);
}

// per-(edge,head) attention logit + per-dst running max
__global__ void k_edge_logits(const float* __restrict__ hs, const float* __restrict__ hd,
                              const int* __restrict__ src, const int* __restrict__ dst,
                              const float* __restrict__ attn,  // [NH*DH]
                              float* __restrict__ logits, unsigned* __restrict__ mmax, int E)
{
    int idx = blockIdx.x * blockDim.x + threadIdx.x;
    if (idx >= E * kNH) return;
    int e = idx / kNH, h = idx % kNH;
    int s = src[e], d = dst[e];
    const float* ps = hs + (size_t)s * kH + h * kDH;
    const float* pd = hd + (size_t)d * kH + h * kDH;
    const float* pa = attn + h * kDH;
    float acc = 0.f;
    #pragma unroll 4
    for (int i = 0; i < kDH; ++i) {
        float v = ps[i] + pd[i];
        v = v > 0.f ? v : 0.2f * v;      // leaky_relu(0.2)
        acc += v * pa[i];
    }
    logits[idx] = acc;
    atomicMax(&mmax[(size_t)d * kNH + h], ford(acc));
}

__global__ void k_edge_exp(const float* __restrict__ logits, const int* __restrict__ dst,
                           const unsigned* __restrict__ mmax,
                           float* __restrict__ aexp, float* __restrict__ z, int E)
{
    int idx = blockIdx.x * blockDim.x + threadIdx.x;
    if (idx >= E * kNH) return;
    int e = idx / kNH, h = idx % kNH;
    int d = dst[e];
    float m = funord(mmax[(size_t)d * kNH + h]);
    float a = __expf(logits[idx] - m);
    aexp[idx] = a;
    atomicAdd(&z[(size_t)d * kNH + h], a);
}

// h[dst] += alpha * hs[src], written into first half of [N,2H] concat buffer
__global__ void k_h_scatter(const float* __restrict__ aexp, const float* __restrict__ z,
                            const float* __restrict__ hs,
                            const int* __restrict__ src, const int* __restrict__ dst,
                            float* __restrict__ out, int ldo, int E)
{
    int idx = blockIdx.x * blockDim.x + threadIdx.x;
    if (idx >= E * kH) return;
    int e = idx >> 8, c = idx & (kH - 1);
    int h = c >> 6;  // c / DH
    int s = src[e], d = dst[e];
    float alpha = aexp[(size_t)e * kNH + h] / z[(size_t)d * kNH + h];
    atomicAdd(&out[(size_t)d * ldo + c], alpha * hs[(size_t)s * kH + c]);
}

// eagg[dst] += em * invdeg[dst], into second half of concat buffer
__global__ void k_eagg_scatter(const float* __restrict__ em, const int* __restrict__ dst,
                               const float* __restrict__ invdeg,
                               float* __restrict__ out, int ldo, int E)
{
    int idx = blockIdx.x * blockDim.x + threadIdx.x;
    if (idx >= E * kH) return;
    int e = idx >> 8, c = idx & (kH - 1);
    int d = dst[e];
    atomicAdd(&out[(size_t)d * ldo + c], em[(size_t)e * kH + c] * invdeg[d]);
}

// wave-per-row LayerNorm(256) + ELU, in place
__global__ void __launch_bounds__(256)
k_layernorm_elu(float* __restrict__ X, int ld,
                const float* __restrict__ g, const float* __restrict__ b, int M)
{
    int row  = blockIdx.x * 8 + (threadIdx.x >> 5);
    int lane = threadIdx.x & 31;
    if (row >= M) return;
    float* p = X + (size_t)row * ld;
    float v[8]; float s = 0.f;
    #pragma unroll
    for (int i = 0; i < 8; ++i) { v[i] = p[lane + 32 * i]; s += v[i]; }
    #pragma unroll
    for (int o = 16; o > 0; o >>= 1) s += __shfl_xor(s, o, 32);
    float mu = s * (1.f / 256.f);
    float var = 0.f;
    #pragma unroll
    for (int i = 0; i < 8; ++i) { float d = v[i] - mu; var += d * d; }
    #pragma unroll
    for (int o = 16; o > 0; o >>= 1) var += __shfl_xor(var, o, 32);
    float rs = rsqrtf(var * (1.f / 256.f) + kEPS);
    #pragma unroll
    for (int i = 0; i < 8; ++i) {
        int c = lane + 32 * i;
        float y = (v[i] - mu) * rs * g[c] + b[c];
        p[c] = y > 0.f ? y : expm1f(y);
    }
}

// column-wise batch stats (sum, sumsq) via per-block partial + atomics
__global__ void k_bn_stats(const float* __restrict__ X, int ld,
                           float* __restrict__ sum, float* __restrict__ sumsq, int M)
{
    int col = threadIdx.x;
    float s = 0.f, ss = 0.f;
    for (int r = blockIdx.x; r < M; r += gridDim.x) {
        float v = X[(size_t)r * ld + col];
        s += v; ss += v * v;
    }
    atomicAdd(&sum[col], s);
    atomicAdd(&sumsq[col], ss);
}

__global__ void k_bn_apply_elu(const float* __restrict__ Xin, int ldi,
                               float* __restrict__ Xout, int ldo,
                               const float* __restrict__ sum, const float* __restrict__ sumsq,
                               const float* __restrict__ g, const float* __restrict__ b, int M)
{
    int idx = blockIdx.x * blockDim.x + threadIdx.x;
    if (idx >= M * kH) return;
    int r = idx >> 8, c = idx & (kH - 1);
    float mean = sum[c] / (float)M;
    float var  = sumsq[c] / (float)M - mean * mean;
    float y = (Xin[(size_t)r * ldi + c] - mean) * rsqrtf(var + kEPS) * g[c] + b[c];
    Xout[(size_t)r * ldo + c] = y > 0.f ? y : expm1f(y);
}

// accumulate per-graph sum (row l*3+2) and max (row l*3+1, uint-mapped) into P[B,9,H]
__global__ void k_graph_stats(const float* __restrict__ X, int ld,
                              const int* __restrict__ n2g, float* __restrict__ P,
                              int l, int Nn)
{
    int idx = blockIdx.x * blockDim.x + threadIdx.x;
    if (idx >= Nn * kH) return;
    int n = idx >> 8, c = idx & (kH - 1);
    float v = X[(size_t)n * ld + c];
    int g = n2g[n];
    size_t base = ((size_t)g * 9 + l * 3) * kH;
    atomicAdd(&P[base + 2 * kH + c], v);
    atomicMax((unsigned*)&P[base + 1 * kH + c], ford(v));
}

__global__ void k_graph_finalize(float* __restrict__ P, const int* __restrict__ gcnt, int l)
{
    int idx = blockIdx.x * blockDim.x + threadIdx.x;
    if (idx >= kB * kH) return;
    int b = idx >> 8, c = idx & (kH - 1);
    size_t base = ((size_t)b * 9 + l * 3) * kH;
    float s = P[base + 2 * kH + c];
    int cnt = gcnt[b]; if (cnt < 1) cnt = 1;
    P[base + c] = s / (float)cnt;                       // mean
    unsigned u = ((unsigned*)P)[base + kH + c];
    P[base + kH + c] = funord(u);                       // max
}

// row dot-product head: out[r] = act(dot(X[r,:K], w) + b0)
template<int ACT>  // 0 none, 3 sigmoid
__global__ void k_rowdot(const float* __restrict__ X, int K,
                         const float* __restrict__ w, const float* __restrict__ b,
                         float* __restrict__ out, int M)
{
    int r = blockIdx.x * blockDim.x + threadIdx.x;
    if (r >= M) return;
    const float* p = X + (size_t)r * K;
    float acc = b[0];
    for (int i = 0; i < K; ++i) acc += p[i] * w[i];
    if (ACT == 3) acc = 1.f / (1.f + __expf(-acc));
    out[r] = acc;
}

__global__ void k_seg_max_node(const float* __restrict__ gl, const int* __restrict__ n2g,
                               unsigned* __restrict__ gmu, int n)
{
    int i = blockIdx.x * blockDim.x + threadIdx.x;
    if (i < n) atomicMax(&gmu[n2g[i]], ford(gl[i]));
}
__global__ void k_gap_exp(const float* __restrict__ gl, const int* __restrict__ n2g,
                          const unsigned* __restrict__ gmu,
                          float* __restrict__ ge, float* __restrict__ gz, int n)
{
    int i = blockIdx.x * blockDim.x + threadIdx.x;
    if (i >= n) return;
    int g = n2g[i];
    float e = __expf(gl[i] - funord(gmu[g]));
    ge[i] = e;
    atomicAdd(&gz[g], e);
}
__global__ void k_gjk_scatter(const float* __restrict__ ge, const float* __restrict__ gz,
                              const int* __restrict__ n2g, const float* __restrict__ xjk,
                              float* __restrict__ gjk, int n)
{
    int idx = blockIdx.x * blockDim.x + threadIdx.x;
    if (idx >= n * kH) return;
    int i = idx >> 8, c = idx & (kH - 1);
    int g = n2g[i];
    atomicAdd(&gjk[(size_t)g * kH + c], (ge[i] / gz[g]) * xjk[(size_t)i * kH + c]);
}

// Y[row, 0:256] = A[row/rowdiv, :], Y[row, 256:512] = Bm[row, :]
__global__ void k_concat2(const float* __restrict__ A, const float* __restrict__ Bm,
                          float* __restrict__ Y, int rows, int rowdiv)
{
    int idx = blockIdx.x * blockDim.x + threadIdx.x;
    if (idx >= rows * 512) return;
    int row = idx >> 9, c = idx & 511;
    float v = (c < 256) ? A[(size_t)(row / rowdiv) * 256 + c]
                        : Bm[(size_t)row * 256 + (c - 256)];
    Y[idx] = v;
}

// fused[b,c] = sum_j gates[b*9+j] * val[(b*9+j), c]
__global__ void k_fsum(const float* __restrict__ gates, const float* __restrict__ val,
                       float* __restrict__ out)
{
    int idx = blockIdx.x * blockDim.x + threadIdx.x;
    if (idx >= kB * kH) return;
    int b = idx >> 8, c = idx & (kH - 1);
    float s = 0.f;
    #pragma unroll
    for (int j = 0; j < 9; ++j)
        s += gates[b * 9 + j] * val[((size_t)(b * 9 + j)) * kH + c];
    out[idx] = s;
}

// ---------------- host orchestration ----------------
extern "C" void kernel_launch(void* const* d_in, const int* in_sizes, int n_in,
                              void* d_out, int out_size, void* d_ws, size_t ws_size,
                              hipStream_t stream)
{
    (void)in_sizes; (void)n_in; (void)out_size; (void)ws_size;
    auto F = [&](int i) { return (const float*)d_in[i]; };
    auto I = [&](int i) { return (const int*)d_in[i]; };

    const float* node_feat = F(0);
    const float* edge_feat = F(1);
    const int*   src       = I(2);
    const int*   dst       = I(3);
    const int*   n2g       = I(4);
    const float* inproj_w  = F(5);
    const float* inproj_b  = F(6);
    // layer params at 7 + 17*l : fc_src_w, fc_src_b, fc_dst_w, fc_dst_b, attn,
    // ec1_w, ec1_b, ec2_w, ec2_b, ef_w, ef_b, ff_w, ff_b, ln_g, ln_b, bn_g, bn_b
    const float* jk_w  = F(58); const float* jk_b  = F(59);
    const float* gap1_w = F(60); const float* gap1_b = F(61);
    const float* gap2_w = F(62); const float* gap2_b = F(63);
    const float* val_w = F(64); const float* val_b = F(65);
    const float* gm1_w = F(66); const float* gm1_b = F(67);
    const float* gm2_w = F(68); const float* gm2_b = F(69);
    const float* po_w  = F(70); const float* po_b  = F(71);
    const float* c1_w  = F(72); const float* c1_b  = F(73);
    const float* c2_w  = F(74); const float* c2_b  = F(75);
    const float* c3_w  = F(76); const float* c3_b  = F(77);

    // workspace layout
    char* wsb = (char*)d_ws;
    size_t off = 0;
    auto alloc = [&](size_t nbytes) -> void* {
        void* p = wsb + off;
        off += (nbytes + 255) & ~(size_t)255;
        return p;
    };
    float* x0   = (float*)alloc((size_t)kN * kH * 4);
    float* hs   = (float*)alloc((size_t)kN * kH * 4);          // later reused as x_jk
    float* hd   = (float*)alloc((size_t)kN * kH * 4);          // later reused as gap hidden
    float* jk   = (float*)alloc((size_t)kN * 3 * kH * 4);      // layer outputs, concat
    float* catb = (float*)alloc((size_t)kN * 2 * kH * 4);      // [h | eagg]
    float* hf   = (float*)alloc((size_t)kN * kH * 4);
    float* es   = (float*)alloc((size_t)kE * kH * 4);
    float* em   = (float*)alloc((size_t)kE * kH * 4);
    float* logits = (float*)alloc((size_t)kE * kNH * 4);
    float* aexp   = (float*)alloc((size_t)kE * kNH * 4);
    unsigned* mmax = (unsigned*)alloc((size_t)kN * kNH * 4);
    float* zbuf  = (float*)alloc((size_t)kN * kNH * 4);
    float* deg   = (float*)alloc((size_t)kN * 4);
    float* invdeg = (float*)alloc((size_t)kN * 4);
    float* bnsum = (float*)alloc(kH * 4);
    float* bnsq  = (float*)alloc(kH * 4);
    int*   gcnt  = (int*)alloc(kB * 4);
    float* P     = (float*)alloc((size_t)kB * 9 * kH * 4);
    float* gl    = (float*)alloc(kN * 4);
    float* ge    = (float*)alloc(kN * 4);
    unsigned* gmu = (unsigned*)alloc(kB * 4);
    float* gz    = (float*)alloc(kB * 4);
    float* gjk   = (float*)alloc((size_t)kB * kH * 4);
    float* cat2  = (float*)alloc((size_t)kB * 9 * 512 * 4);
    float* gh1   = (float*)alloc((size_t)kB * 9 * kH * 4);
    float* gates = (float*)alloc((size_t)kB * 9 * 4);
    float* gval  = (float*)alloc((size_t)kB * 9 * kH * 4);
    float* fsum  = (float*)alloc((size_t)kB * kH * 4);
    float* fused = (float*)alloc((size_t)kB * kH * 4);
    float* emb   = (float*)alloc((size_t)kB * 512 * 4);
    float* ch1   = (float*)alloc((size_t)kB * kH * 4);
    float* ch2   = (float*)alloc((size_t)kB * 128 * 4);

    const int T = 256;
    auto blk = [&](long n) { return (unsigned)((n + T - 1) / T); };

    // degrees + graph counts
    hipMemsetAsync(deg, 0, kN * 4, stream);
    k_count_deg<<<blk(kE), T, 0, stream>>>(dst, deg, kE);
    k_invdeg<<<blk(kN), T, 0, stream>>>(deg, invdeg, kN);
    hipMemsetAsync(gcnt, 0, kB * 4, stream);
    k_count_int<<<blk(kN), T, 0, stream>>>(n2g, gcnt, kN);

    // input projection + ELU
    gemm(1, node_feat, kIN, inproj_w, inproj_b, x0, kH, kN, kIN, kH, stream);

    hipMemsetAsync(P, 0, (size_t)kB * 9 * kH * 4, stream);

    const float* xin = x0;
    int ldxin = kH;
    for (int l = 0; l < kL; ++l) {
        int pb = 7 + 17 * l;
        const float* fc_src_w = F(pb + 0); const float* fc_src_b = F(pb + 1);
        const float* fc_dst_w = F(pb + 2); const float* fc_dst_b = F(pb + 3);
        const float* attn     = F(pb + 4);
        const float* ec1_w = F(pb + 5); const float* ec1_b = F(pb + 6);
        const float* ec2_w = F(pb + 7); const float* ec2_b = F(pb + 8);
        const float* ef_w  = F(pb + 9); const float* ef_b  = F(pb + 10);
        const float* ff_w  = F(pb + 11); const float* ff_b = F(pb + 12);
        const float* ln_g  = F(pb + 13); const float* ln_b = F(pb + 14);
        const float* bn_g  = F(pb + 15); const float* bn_b = F(pb + 16);

        // GATv2: projections
        gemm(0, xin, ldxin, fc_src_w, fc_src_b, hs, kH, kN, kH, kH, stream);
        gemm(0, xin, ldxin, fc_dst_w, fc_dst_b, hd, kH, kN, kH, kH, stream);

        // attention softmax over in-edges of each dst
        hipMemsetAsync(mmax, 0, (size_t)kN * kNH * 4, stream);
        k_edge_logits<<<blk((long)kE * kNH), T, 0, stream>>>(hs, hd, src, dst, attn,
                                                             logits, mmax, kE);
        hipMemsetAsync(zbuf, 0, (size_t)kN * kNH * 4, stream);
        k_edge_exp<<<blk((long)kE * kNH), T, 0, stream>>>(logits, dst, mmax, aexp, zbuf, kE);

        hipMemsetAsync(catb, 0, (size_t)kN * 2 * kH * 4, stream);
        k_h_scatter<<<blk((long)kE * kH), T, 0, stream>>>(aexp, zbuf, hs, src, dst,
                                                          catb, 2 * kH, kE);

        // multi-scale edge conv
        gemm(0, edge_feat, kED, ec1_w, ec1_b, es, kH, kE, kED, 128, stream);
        gemm(0, edge_feat, kED, ec2_w, ec2_b, es + 128, kH, kE, kED, 128, stream);
        gemm(1, es, kH, ef_w, ef_b, em, kH, kE, kH, kH, stream);
        k_eagg_scatter<<<blk((long)kE * kH), T, 0, stream>>>(em, dst, invdeg,
                                                             catb + kH, 2 * kH, kE);

        // fusion linear + LN + ELU + BN + ELU
        gemm(0, catb, 2 * kH, ff_w, ff_b, hf, kH, kN, 2 * kH, kH, stream);
        k_layernorm_elu<<<(kN + 7) / 8, T, 0, stream>>>(hf, kH, ln_g, ln_b, kN);

        hipMemsetAsync(bnsum, 0, kH * 4, stream);
        hipMemsetAsync(bnsq, 0, kH * 4, stream);
        k_bn_stats<<<512, T, 0, stream>>>(hf, kH, bnsum, bnsq, kN);

        float* xout = jk + (size_t)l * kH;            // slice of [N,3H]
        k_bn_apply_elu<<<blk((long)kN * kH), T, 0, stream>>>(hf, kH, xout, 3 * kH,
                                                             bnsum, bnsq, bn_g, bn_b, kN);
        // per-graph mean/max/sum into P
        k_graph_stats<<<blk((long)kN * kH), T, 0, stream>>>(xout, 3 * kH, n2g, P, l, kN);
        k_graph_finalize<<<blk((long)kB * kH), T, 0, stream>>>(P, gcnt, l);

        xin = xout; ldxin = 3 * kH;
    }

    // jumping knowledge
    float* xjk = hs;   // reuse
    gemm(0, jk, 3 * kH, jk_w, jk_b, xjk, kH, kN, 3 * kH, kH, stream);

    // global attention pooling
    float* gh = hd;    // reuse, [N,128]
    gemm(2, xjk, kH, gap1_w, gap1_b, gh, 128, kN, kH, 128, stream);
    k_rowdot<0><<<blk(kN), T, 0, stream>>>(gh, 128, gap2_w, gap2_b, gl, kN);
    hipMemsetAsync(gmu, 0, kB * 4, stream);
    k_seg_max_node<<<blk(kN), T, 0, stream>>>(gl, n2g, gmu, kN);
    hipMemsetAsync(gz, 0, kB * 4, stream);
    k_gap_exp<<<blk(kN), T, 0, stream>>>(gl, n2g, gmu, ge, gz, kN);
    hipMemsetAsync(gjk, 0, (size_t)kB * kH * 4, stream);
    k_gjk_scatter<<<blk((long)kN * kH), T, 0, stream>>>(ge, gz, n2g, xjk, gjk, kN);

    // gated multiscale fusion
    k_concat2<<<blk((long)kB * 9 * 512), T, 0, stream>>>(gjk, P, cat2, kB * 9, 9);
    gemm(2, cat2, 512, gm1_w, gm1_b, gh1, kH, kB * 9, 512, kH, stream);
    k_rowdot<3><<<blk(kB * 9), T, 0, stream>>>(gh1, kH, gm2_w, gm2_b, gates, kB * 9);
    gemm(0, P, kH, val_w, val_b, gval, kH, kB * 9, kH, kH, stream);
    k_fsum<<<blk((long)kB * kH), T, 0, stream>>>(gates, gval, fsum);
    gemm(0, fsum, kH, po_w, po_b, fused, kH, kB, kH, kH, stream);

    // classifier
    k_concat2<<<blk((long)kB * 512), T, 0, stream>>>(gjk, fused, emb, kB, 1);
    gemm(1, emb, 512, c1_w, c1_b, ch1, kH, kB, 512, kH, stream);
    gemm(1, ch1, kH, c2_w, c2_b, ch2, 128, kB, kH, 128, stream);
    k_rowdot<0><<<blk(kB), T, 0, stream>>>(ch2, 128, c3_w, c3_b, (float*)d_out, kB);
}
